// CrystalMPNN_71829033058346
// MI455X (gfx1250) — compile-verified, matrix operand order checked by reading, not run on hardware
//
#include <hip/hip_runtime.h>
#include <math.h>
#include <stdint.h>

#define NN 50000
#define EE 600000
#define GG 512
#define HH 128
#define LL 4

typedef __bf16 bf16_t;
typedef __attribute__((ext_vector_type(16))) __bf16 bf16x16;
typedef __attribute__((ext_vector_type(8)))  __bf16 bf16x8;
typedef __attribute__((ext_vector_type(8)))  float  floatx8;

// ---------------------------------------------------------------------------
// CDNA5 async global->LDS copy (tracked by ASYNCcnt). lds_off is the LDS byte
// offset (low 32 bits of the generic shared-pointer address on AMDGPU).
// ---------------------------------------------------------------------------
__device__ __forceinline__ void async_load_b128_to_lds(uint32_t lds_off,
                                                       const void* gaddr)
{
    asm volatile("global_load_async_to_lds_b128 %0, %1, off"
                 :
                 : "v"(lds_off), "v"(gaddr)
                 : "memory");
}
__device__ __forceinline__ void wait_asynccnt0()
{
    asm volatile("s_wait_asynccnt 0x0" ::: "memory");
}

// ---------------------------------------------------------------------------
// Pack a [K,128] f32 weight matrix into bf16, laid out in WMMA B-fragment
// order for v_wmma_f32_16x16x32_bf16:
//   offset = ((kb*8 + ntile)*32 + lane)*16 + i
// lane carries column N = ntile*16 + (lane&15); element i carries
//   K = kb*32 + ((lane>=16)?8:0) + (i<8 ? i : 16+(i-8))
// so each wave fragment load is one contiguous 32-byte read per lane.
// ---------------------------------------------------------------------------
__global__ void pack_weight_kernel(const float* __restrict__ W,
                                   bf16_t* __restrict__ out, int K)
{
    int tid = blockIdx.x * blockDim.x + threadIdx.x;
    if (tid >= K * HH) return;
    int i    =  tid        & 15;
    int lane = (tid >> 4)  & 31;
    int nt   = (tid >> 9)  & 7;
    int kb   =  tid >> 12;
    int kk   = ((lane >> 4) << 3) + ((i < 8) ? i : (16 + (i - 8)));
    int k    = kb * 32 + kk;
    int col  = nt * 16 + (lane & 15);
    out[tid] = (bf16_t)W[(size_t)k * HH + col];
}

// First encoder layers (K = 7 or 2): plain VALU, outputs relu(x@W+b) as bf16.
__global__ void encode_l1_kernel(const float* __restrict__ X, int Kin,
                                 const float* __restrict__ W,
                                 const float* __restrict__ b,
                                 bf16_t* __restrict__ out, long long R)
{
    long long tid = (long long)blockIdx.x * blockDim.x + threadIdx.x;
    if (tid >= R * HH) return;
    int j = (int)(tid & (HH - 1));
    long long r = tid >> 7;
    float acc = b[j];
    for (int k = 0; k < Kin; ++k)
        acc = fmaf(X[r * Kin + k], W[(size_t)k * HH + j], acc);
    out[tid] = (bf16_t)fmaxf(acc, 0.0f);
}

#define MODE_PLAIN 0
#define MODE_MSG   1
#define MODE_UPD   2

// ---------------------------------------------------------------------------
// Generic wave-tile GEMM: each wave computes 16x128 tiles of C = A@B + bias.
//   MODE_PLAIN: A = A0 [R,128] bf16 row-major          (KB = 4)
//   MODE_MSG  : A = [ h[src] | h[dst] | e ], K = 384   (KB = 12, fused gather)
//   MODE_UPD  : A = [ h | h_agg ], K = 256             (KB = 8)
// MODE and KB are template params: the K-loop fully unrolls, A-source
// selection is compile-time, and per K-step all 8 B fragments are loaded
// from LDS in one clause (single s_wait_dscnt) before 8 back-to-back WMMAs.
// The packed weight panel is staged once per workgroup into LDS with CDNA5
// async global->LDS copies (ASYNCcnt). Grid-stride amortizes the staging.
// Optional epilogue: ReLU, residual, f32/bf16 stores, atomic scatter-add.
// ---------------------------------------------------------------------------
template <int MODE, int KB>
__global__ __launch_bounds__(256) void gemm128_kernel(
    int R,
    const bf16_t* __restrict__ A0, const bf16_t* __restrict__ A1,
    const bf16_t* __restrict__ A2,
    const int* __restrict__ srcIdx, const int* __restrict__ dstIdx,
    const bf16_t* __restrict__ Bp, const float* __restrict__ bias,
    int do_relu,
    float* Cf, bf16_t* Cb, const float* Resid,
    const int* __restrict__ scatIdx, float* scatOut)
{
    extern __shared__ char smem_raw[];
    bf16_t* Bs = (bf16_t*)smem_raw;                 // KB*4096 bf16

    // --- stage packed weights into LDS (async, 16B per lane per issue) ---
    // KB*512 16-byte chunks; always a multiple of 256 -> no divergence.
    for (int c = threadIdx.x; c < KB * 512; c += blockDim.x) {
        uint32_t lds_off = (uint32_t)(uintptr_t)(Bs + (size_t)c * 8);
        async_load_b128_to_lds(lds_off, (const void*)(Bp + (size_t)c * 8));
    }
    wait_asynccnt0();
    __syncthreads();

    const int lane   = threadIdx.x & 31;
    const int wave   = threadIdx.x >> 5;
    const int nwaves = blockDim.x >> 5;
    const int tiles  = R >> 4;                      // R is a multiple of 16
    const int rlo    = lane & 15;
    const int khalf  = (lane >> 4) << 3;            // 0 (lanes 0-15) or 8

    for (int tile = blockIdx.x * nwaves + wave; tile < tiles;
         tile += gridDim.x * nwaves) {

        const int r = tile * 16 + rlo;
        int gsrc = r, gdst = r;
        if (MODE == MODE_MSG) { gsrc = srcIdx[r]; gdst = dstIdx[r]; }

        floatx8 acc[8];
#pragma unroll
        for (int t = 0; t < 8; ++t)
#pragma unroll
            for (int q = 0; q < 8; ++q) acc[t][q] = 0.0f;

#pragma unroll
        for (int kb = 0; kb < KB; ++kb) {
            const bf16_t* arow;
            int koff;
            if (MODE == MODE_MSG) {
                if      (kb < 4) { arow = A0 + (size_t)gsrc * HH; koff = kb * 32; }
                else if (kb < 8) { arow = A0 + (size_t)gdst * HH; koff = (kb - 4) * 32; }
                else             { arow = A2 + (size_t)r    * HH; koff = (kb - 8) * 32; }
            } else if (MODE == MODE_UPD) {
                if (kb < 4) { arow = A0 + (size_t)r * HH; koff = kb * 32; }
                else        { arow = A1 + (size_t)r * HH; koff = (kb - 4) * 32; }
            } else {
                arow = A0 + (size_t)r * HH; koff = kb * 32;
            }
            bf16x8 lo = *(const bf16x8*)(arow + koff + khalf);
            bf16x8 hi = *(const bf16x8*)(arow + koff + khalf + 16);
            bf16x16 afrag = __builtin_shufflevector(lo, hi,
                0,1,2,3,4,5,6,7,8,9,10,11,12,13,14,15);

            // Load all 8 B fragments for this K-step, then 8 WMMAs.
            const bf16_t* bbase = Bs + (size_t)kb * 4096 + (size_t)lane * 16;
            bf16x16 bfr[8];
#pragma unroll
            for (int nt = 0; nt < 8; ++nt)
                bfr[nt] = *(const bf16x16*)(bbase + (size_t)nt * 512);
#pragma unroll
            for (int nt = 0; nt < 8; ++nt)
                acc[nt] = __builtin_amdgcn_wmma_f32_16x16x32_bf16(
                    false, afrag, false, bfr[nt], (short)0, acc[nt], false, false);
        }

        // C/D layout: lane 0-15 -> M = v, lane 16-31 -> M = 8+v; N = lane&15
        int rows[8];
#pragma unroll
        for (int v = 0; v < 8; ++v) rows[v] = tile * 16 + khalf + v;
        int sidx[8];
        if (scatOut) {
#pragma unroll
            for (int v = 0; v < 8; ++v) sidx[v] = scatIdx[rows[v]];
        }

#pragma unroll
        for (int nt = 0; nt < 8; ++nt) {
            int col  = nt * 16 + rlo;
            float bv = bias[col];
#pragma unroll
            for (int v = 0; v < 8; ++v) {
                float val = acc[nt][v] + bv;
                if (do_relu) val = fmaxf(val, 0.0f);
                size_t idx = (size_t)rows[v] * HH + col;
                if (Resid)   val += Resid[idx];
                if (Cf)      Cf[idx] = val;
                if (Cb)      Cb[idx] = (bf16_t)val;
                if (scatOut) atomicAdd(&scatOut[(size_t)sidx[v] * HH + col], val);
            }
        }
    }
}

__global__ void fillf_kernel(float* p, float v, long long n)
{
    long long tid = (long long)blockIdx.x * blockDim.x + threadIdx.x;
    if (tid < n) p[tid] = v;
}

__global__ void f2bf_kernel(const float* __restrict__ in,
                            bf16_t* __restrict__ out, long long n)
{
    long long tid = (long long)blockIdx.x * blockDim.x + threadIdx.x;
    if (tid < n) out[tid] = (bf16_t)in[tid];
}

// gate[n] = hidden[n,:] @ gw2 + gb2
__global__ void gate_dot_kernel(const bf16_t* __restrict__ hid,
                                const float* __restrict__ w,
                                const float* __restrict__ b,
                                float* __restrict__ gate, int n)
{
    int i = blockIdx.x * blockDim.x + threadIdx.x;
    if (i >= n) return;
    float acc = b[0];
    const bf16_t* row = hid + (size_t)i * HH;
    for (int j = 0; j < HH; ++j) acc = fmaf((float)row[j], w[j], acc);
    gate[i] = acc;
}

__device__ __forceinline__ void atomic_max_f32(float* addr, float val)
{
    unsigned int* ua = (unsigned int*)addr;
    unsigned int old = *ua;
    while (true) {
        float cur = __uint_as_float(old);
        if (cur >= val) break;
        unsigned int assumed = old;
        old = atomicCAS(ua, assumed, __float_as_uint(val));
        if (old == assumed) break;
    }
}

__global__ void segmax_kernel(const float* __restrict__ gate,
                              const int* __restrict__ gid,
                              float* gmax, int n)
{
    int i = blockIdx.x * blockDim.x + threadIdx.x;
    if (i >= n) return;
    atomic_max_f32(&gmax[gid[i]], gate[i]);
}

__global__ void exden_kernel(const float* __restrict__ gate,
                             const int* __restrict__ gid,
                             const float* __restrict__ gmax,
                             float* __restrict__ ex, float* den, int n)
{
    int i = blockIdx.x * blockDim.x + threadIdx.x;
    if (i >= n) return;
    int g = gid[i];
    float e = expf(gate[i] - gmax[g]);
    ex[i] = e;
    atomicAdd(&den[g], e);
}

__global__ void emb_kernel(const float* __restrict__ ex,
                           const float* __restrict__ den,
                           const int* __restrict__ gid,
                           const float* __restrict__ h,
                           float* emb, long long n128)
{
    long long tid = (long long)blockIdx.x * blockDim.x + threadIdx.x;
    if (tid >= n128) return;
    long long nidx = tid >> 7;
    int j = (int)(tid & (HH - 1));
    int g = gid[nidx];
    float alpha = ex[nidx] / den[g];
    atomicAdd(&emb[(size_t)g * HH + j], alpha * h[tid]);
}

__global__ void heads_kernel(const float* __restrict__ emb,
    const float* w0, const float* b0, const float* w1, const float* b1,
    const float* w2, const float* b2, const float* w3, const float* b3,
    float* o0, float* o1, float* o2, float* o3)
{
    int g = blockIdx.x * blockDim.x + threadIdx.x;
    if (g >= GG) return;
    float a0 = b0[0], a1 = b1[0], a2 = b2[0], a3 = b3[0];
    const float* e = emb + (size_t)g * HH;
    for (int j = 0; j < HH; ++j) {
        float x = e[j];
        a0 = fmaf(x, w0[j], a0); a1 = fmaf(x, w1[j], a1);
        a2 = fmaf(x, w2[j], a2); a3 = fmaf(x, w3[j], a3);
    }
    o0[g] = a0; o1[g] = a1; o2[g] = a2; o3[g] = a3;
}

// ---------------------------------------------------------------------------
extern "C" void kernel_launch(void* const* d_in, const int* in_sizes, int n_in,
                              void* d_out, int out_size, void* d_ws, size_t ws_size,
                              hipStream_t stream)
{
    const float* node_feat = (const float*)d_in[0];
    const float* edge_feat = (const float*)d_in[1];
    const int*   src  = (const int*)d_in[2];
    const int*   dst  = (const int*)d_in[3];
    const int*   gid  = (const int*)d_in[4];
    const float *nw1 = (const float*)d_in[5],  *nb1 = (const float*)d_in[6];
    const float *nw2 = (const float*)d_in[7],  *nb2 = (const float*)d_in[8];
    const float *ew1 = (const float*)d_in[9],  *eb1 = (const float*)d_in[10];
    const float *ew2 = (const float*)d_in[11], *eb2 = (const float*)d_in[12];
    const float *mw1 = (const float*)d_in[13], *mb1 = (const float*)d_in[14];
    const float *mw2 = (const float*)d_in[15], *mb2 = (const float*)d_in[16];
    const float *uw1 = (const float*)d_in[17], *ub1 = (const float*)d_in[18];
    const float *uw2 = (const float*)d_in[19], *ub2 = (const float*)d_in[20];
    const float *gw1 = (const float*)d_in[21], *gb1 = (const float*)d_in[22];
    const float *gw2 = (const float*)d_in[23], *gb2 = (const float*)d_in[24];
    const float *hw_bg = (const float*)d_in[25], *hb_bg = (const float*)d_in[26];
    const float *hw_fe = (const float*)d_in[27], *hb_fe = (const float*)d_in[28];
    const float *hw_st = (const float*)d_in[29], *hb_st = (const float*)d_in[30];
    const float *hw_de = (const float*)d_in[31], *hb_de = (const float*)d_in[32];

    char* ws = (char*)d_ws;
    size_t off = 0;
    auto alloc = [&](size_t bytes) {
        size_t o = off; off += (bytes + 255) & ~(size_t)255; return o;
    };

    bf16_t* pw_nw2 = (bf16_t*)(ws + alloc((size_t)HH * HH * 2));
    bf16_t* pw_ew2 = (bf16_t*)(ws + alloc((size_t)HH * HH * 2));
    bf16_t* pw_gw1 = (bf16_t*)(ws + alloc((size_t)HH * HH * 2));
    bf16_t* pw_mw1 = (bf16_t*)(ws + alloc((size_t)LL * 384 * HH * 2));
    bf16_t* pw_mw2 = (bf16_t*)(ws + alloc((size_t)LL * 128 * HH * 2));
    bf16_t* pw_uw1 = (bf16_t*)(ws + alloc((size_t)LL * 256 * HH * 2));
    bf16_t* pw_uw2 = (bf16_t*)(ws + alloc((size_t)LL * 128 * HH * 2));
    bf16_t* hbf    = (bf16_t*)(ws + alloc((size_t)NN * HH * 2));
    bf16_t* haggbf = (bf16_t*)(ws + alloc((size_t)NN * HH * 2));
    bf16_t* u1     = (bf16_t*)(ws + alloc((size_t)NN * HH * 2)); // encoder-t / gate hidden
    bf16_t* ebf    = (bf16_t*)(ws + alloc((size_t)EE * HH * 2));
    bf16_t* m1     = (bf16_t*)(ws + alloc((size_t)EE * HH * 2)); // also edge-encoder t
    float*  haggf  = (float*)(ws + alloc((size_t)NN * HH * 4));
    float*  gatef  = (float*)(ws + alloc((size_t)NN * 4));
    float*  exb    = (float*)(ws + alloc((size_t)NN * 4));
    float*  gmax   = (float*)(ws + alloc((size_t)GG * 4));
    float*  den    = (float*)(ws + alloc((size_t)GG * 4));

    float* h_out  = (float*)d_out;           // h  [N,H]  (output 0, residual buf)
    float* emb    = h_out + (size_t)NN * HH; // emb [G,H]
    float* bg     = emb + (size_t)GG * HH;
    float* fe     = bg + GG;
    float* st     = fe + GG;
    float* de     = st + GG;

    auto pack = [&](const float* W, bf16_t* out, int K) {
        int total = K * HH;
        pack_weight_kernel<<<(total + 255) / 256, 256, 0, stream>>>(W, out, K);
    };
    auto gridFor = [&](int R) {
        int tiles = R / 16;
        int blocks = (tiles + 7) / 8;
        return blocks > 1024 ? 1024 : blocks;   // grid-stride; amortize staging
    };
    auto fillf = [&](float* p, float v, long long n) {
        fillf_kernel<<<(unsigned)((n + 255) / 256), 256, 0, stream>>>(p, v, n);
    };

    // --- pack all H-width weights into WMMA fragment order (bf16) ---
    pack(nw2, pw_nw2, 128);
    pack(ew2, pw_ew2, 128);
    pack(gw1, pw_gw1, 128);
    for (int l = 0; l < LL; ++l) {
        pack(mw1 + (size_t)l * 384 * HH, pw_mw1 + (size_t)l * 384 * HH, 384);
        pack(mw2 + (size_t)l * 128 * HH, pw_mw2 + (size_t)l * 128 * HH, 128);
        pack(uw1 + (size_t)l * 256 * HH, pw_uw1 + (size_t)l * 256 * HH, 256);
        pack(uw2 + (size_t)l * 128 * HH, pw_uw2 + (size_t)l * 128 * HH, 128);
    }

    // --- node encoder: h = relu(nf@nw1+nb1) @ nw2 + nb2 ---
    encode_l1_kernel<<<(unsigned)(((long long)NN * HH + 255) / 256), 256, 0, stream>>>(
        node_feat, 7, nw1, nb1, u1, NN);
    gemm128_kernel<MODE_PLAIN, 4><<<gridFor(NN), 256, 4 * 8192, stream>>>(
        NN, u1, nullptr, nullptr, nullptr, nullptr,
        pw_nw2, nb2, 0, h_out, hbf, nullptr, nullptr, nullptr);

    // --- edge encoder: e = relu(ef@ew1+eb1) @ ew2 + eb2 (bf16 only) ---
    encode_l1_kernel<<<(unsigned)(((long long)EE * HH + 255) / 256), 256, 0, stream>>>(
        edge_feat, 2, ew1, eb1, m1, EE);
    gemm128_kernel<MODE_PLAIN, 4><<<gridFor(EE), 256, 4 * 8192, stream>>>(
        EE, m1, nullptr, nullptr, nullptr, nullptr,
        pw_ew2, eb2, 0, nullptr, ebf, nullptr, nullptr, nullptr);

    // --- message-passing layers ---
    for (int l = 0; l < LL; ++l) {
        fillf(haggf, 0.0f, (long long)NN * HH);
        // m1 = relu([h[src]|h[dst]|e] @ mw1 + mb1)   (K=384, fused gather)
        gemm128_kernel<MODE_MSG, 12><<<gridFor(EE), 256, 12 * 8192, stream>>>(
            EE, hbf, nullptr, ebf, src, dst,
            pw_mw1 + (size_t)l * 384 * HH, mb1 + (size_t)l * HH, 1,
            nullptr, m1, nullptr, nullptr, nullptr);
        // h_agg[dst] += m1 @ mw2 + mb2               (fused scatter-sum)
        gemm128_kernel<MODE_PLAIN, 4><<<gridFor(EE), 256, 4 * 8192, stream>>>(
            EE, m1, nullptr, nullptr, nullptr, nullptr,
            pw_mw2 + (size_t)l * 128 * HH, mb2 + (size_t)l * HH, 0,
            nullptr, nullptr, nullptr, dst, haggf);
        f2bf_kernel<<<(unsigned)(((long long)NN * HH + 255) / 256), 256, 0, stream>>>(
            haggf, haggbf, (long long)NN * HH);
        // u1 = relu([h|h_agg] @ uw1 + ub1)           (K=256)
        gemm128_kernel<MODE_UPD, 8><<<gridFor(NN), 256, 8 * 8192, stream>>>(
            NN, hbf, haggbf, nullptr, nullptr, nullptr,
            pw_uw1 + (size_t)l * 256 * HH, ub1 + (size_t)l * HH, 1,
            nullptr, u1, nullptr, nullptr, nullptr);
        // h = u1 @ uw2 + ub2 + h (residual); refresh both f32 and bf16 copies
        gemm128_kernel<MODE_PLAIN, 4><<<gridFor(NN), 256, 4 * 8192, stream>>>(
            NN, u1, nullptr, nullptr, nullptr, nullptr,
            pw_uw2 + (size_t)l * 128 * HH, ub2 + (size_t)l * HH, 0,
            h_out, hbf, h_out, nullptr, nullptr);
    }

    // --- global attention pooling ---
    gemm128_kernel<MODE_PLAIN, 4><<<gridFor(NN), 256, 4 * 8192, stream>>>(
        NN, hbf, nullptr, nullptr, nullptr, nullptr,
        pw_gw1, gb1, 1, nullptr, u1, nullptr, nullptr, nullptr);
    gate_dot_kernel<<<(NN + 255) / 256, 256, 0, stream>>>(u1, gw2, gb2, gatef, NN);
    fillf(gmax, -3.0e38f, GG);
    fillf(den, 0.0f, GG);
    fillf(emb, 0.0f, (long long)GG * HH);
    segmax_kernel<<<(NN + 255) / 256, 256, 0, stream>>>(gatef, gid, gmax, NN);
    exden_kernel<<<(NN + 255) / 256, 256, 0, stream>>>(gatef, gid, gmax, exb, den, NN);
    emb_kernel<<<(unsigned)(((long long)NN * HH + 255) / 256), 256, 0, stream>>>(
        exb, den, gid, h_out, emb, (long long)NN * HH);

    // --- property heads ---
    heads_kernel<<<(GG + 255) / 256, 256, 0, stream>>>(
        emb, hw_bg, hb_bg, hw_fe, hb_fe, hw_st, hb_st, hw_de, hb_de,
        bg, fe, st, de);
}
